// PWCNet_60009283060017
// MI455X (gfx1250) — matrix operand order repaired and checked by reading, not compiled, not run on hardware
//
#include <hip/hip_runtime.h>

typedef __attribute__((ext_vector_type(16))) _Float16 v16h;
typedef __attribute__((ext_vector_type(8)))  _Float16 v8h;
typedef __attribute__((ext_vector_type(8)))  float    v8f;

#define LRELU_NEG 0.1f

// ---------------------------------------------------------------------------
// fp32 -> fp16 weight conversion
// ---------------------------------------------------------------------------
__global__ void pwc_f32_to_f16(const float* __restrict__ src, _Float16* __restrict__ dst, int n) {
    int i = blockIdx.x * 256 + threadIdx.x;
    if (i < n) dst[i] = (_Float16)src[i];
}

// ---------------------------------------------------------------------------
// Implicit-GEMM 3x3 conv via v_wmma_f32_16x16x32_f16.
//   Grid: x = Wout/32 tiles (32 consecutive output pixels along W),
//         y = Cout/32 tiles, z = B*Hout (one output row per z).
//   128 threads = 4 waves; each wave computes one 16x16 tile of a 32Mx32N block.
//   A staged [m][k] (row stride 40 halves, 16B-aligned runs),
//   B staged transposed [n][k] so lane fragments are contiguous -> ds_load_b128.
//   Staging is BRANCHLESS: addresses are clamped in-bounds, OOB lanes select 0,
//   so global loads pipeline without exec-mask branches or per-load waits.
//   Weights prepacked fp16 HWIO: wt[((ky*3+kx)*Cin + ci)*Cout + co].
// ---------------------------------------------------------------------------
__global__ __launch_bounds__(128)
void pwc_conv3x3_wmma(const float* __restrict__ in, const _Float16* __restrict__ wt,
                      const float* __restrict__ bias, float* __restrict__ out,
                      int B, int Hin, int Win, int Cin,
                      int Hout, int Wout, int Cout,
                      int stride, int dil, int pad, int relu) {
    const int tidx  = threadIdx.x;
    const int lane  = tidx & 31;
    const int wave  = tidx >> 5;
    const int wm    = wave >> 1;            // 16-row half of the 32-pixel tile
    const int wn    = wave & 1;             // 16-col half of the 32-cout tile
    const int cA    = tidx & 31;            // staged channel (A) / cout (B) column
    const int rBase = tidx >> 5;            // staged row base (step 4)

    const int z  = blockIdx.z;              // b*Hout + oy
    const int b  = z / Hout;
    const int oy = z - b * Hout;
    const int ox0 = blockIdx.x * 32;
    const int n0  = blockIdx.y * 32;

    const int nB     = n0 + cA;             // cout column this thread stages
    const bool nOK   = nB < Cout;
    const int nSafe  = nOK ? nB : (Cout - 1);

    __shared__ __align__(16) _Float16 lA[32][40];   // [m][k], 80B row stride
    __shared__ __align__(16) _Float16 lBt[32][40];  // [n][k] (transposed)

    v8f acc = {};

    const int kcTiles = (Cin + 31) >> 5;

    for (int ky = 0; ky < 3; ++ky) {
        const int iy = oy * stride - pad + ky * dil;
        const bool iyOK = (iy >= 0) & (iy < Hin);
        const int iyC = iy < 0 ? 0 : (iy >= Hin ? Hin - 1 : iy);
        const size_t rowBase = ((size_t)b * Hin + iyC) * Win;
        for (int kx = 0; kx < 3; ++kx) {
            const int kxd = kx * dil - pad;
            const int tap = ky * 3 + kx;

            // per-tap clamped pixel pointers + validity masks (kc-invariant)
            const float* aptr[8];
            bool aok[8];
#pragma unroll
            for (int i = 0; i < 8; ++i) {
                int r  = rBase + (i << 2);
                int ox = ox0 + r;
                int ix = ox * stride + kxd;
                aok[i] = iyOK & (ox < Wout) & (ix >= 0) & (ix < Win);
                int ixC = ix < 0 ? 0 : (ix >= Win ? Win - 1 : ix);
                aptr[i] = in + (rowBase + ixC) * Cin;
            }
            const _Float16* wtap = wt + (size_t)tap * Cin * Cout;

            for (int kc = 0; kc < kcTiles; ++kc) {
                const int kc0 = kc << 5;
                // ---- stage A tile: 32 pixels (along W) x 32 channels ----
                {
                    const int ch   = kc0 + cA;
                    const bool chOK = ch < Cin;
                    const int chS  = chOK ? ch : (Cin - 1);
                    float va[8];
#pragma unroll
                    for (int i = 0; i < 8; ++i)
                        va[i] = aptr[i][chS];            // always-safe load
#pragma unroll
                    for (int i = 0; i < 8; ++i) {
                        float v = (aok[i] & chOK) ? va[i] : 0.0f;
                        lA[rBase + (i << 2)][cA] = (_Float16)v;
                    }
                }
                // ---- stage B tile transposed: lBt[n][k] ----
                {
                    _Float16 vb[8];
                    bool bok[8];
#pragma unroll
                    for (int i = 0; i < 8; ++i) {
                        int r  = rBase + (i << 2);       // k index within chunk
                        int ch = kc0 + r;
                        bok[i] = nOK & (ch < Cin);
                        int chS = ch < Cin ? ch : (Cin - 1);
                        vb[i] = wtap[(size_t)chS * Cout + nSafe];   // always-safe load
                    }
#pragma unroll
                    for (int i = 0; i < 8; ++i) {
                        _Float16 v = bok[i] ? vb[i] : (_Float16)0.0f;
                        lBt[cA][rBase + (i << 2)] = v;
                    }
                }
                __syncthreads();

                // ---- per-lane fragments: contiguous 16B runs -> ds_load_b128 ----
                const int halfSel = lane >> 4;           // 0: k 0-7,16-23; 1: k 8-15,24-31
                const int mrow = (wm << 4) + (lane & 15);
                const int nrow = (wn << 4) + (lane & 15);
                const v8h a_lo = *(const v8h*)&lA[mrow][halfSel << 3];         // k: 8*sel..+7
                const v8h a_hi = *(const v8h*)&lA[mrow][16 + (halfSel << 3)];  // k: 16+8*sel..
                const v8h b_lo = *(const v8h*)&lBt[nrow][halfSel << 3];        // k: 8*sel..+7
                const v8h b_hi = *(const v8h*)&lBt[nrow][16 + (halfSel << 3)]; // k: 16+8*sel..
                v16h afrag = __builtin_shufflevector(a_lo, a_hi,
                        0,1,2,3,4,5,6,7,8,9,10,11,12,13,14,15);
                v16h bfrag = __builtin_shufflevector(b_lo, b_hi,
                        0,1,2,3,4,5,6,7,8,9,10,11,12,13,14,15);

                acc = __builtin_amdgcn_wmma_f32_16x16x32_f16(
                        false, afrag, false, bfrag, (short)0, acc, false, false);
                __syncthreads();
            }
        }
    }

    // ---- epilogue: bias + leaky relu, scatter per C/D layout ----
    const int n = n0 + (wn << 4) + (lane & 15);
    const int oxw = ox0 + (wm << 4) + ((lane >> 4) << 3);
    if (n < Cout) {
        const float bb = bias[n];
        const size_t outRow = ((size_t)b * Hout + oy) * Wout;
#pragma unroll
        for (int r = 0; r < 8; ++r) {
            int ox = oxw + r;
            if (ox < Wout) {
                float v = acc[r] + bb;
                if (relu) v = v > 0.0f ? v : LRELU_NEG * v;
                out[(outRow + ox) * (size_t)Cout + n] = v;
            }
        }
    }
}

// ---------------------------------------------------------------------------
// Direct 3x3 conv (small Cout: flow heads). Optional residual add into out.
// ---------------------------------------------------------------------------
__global__ void pwc_conv3x3_direct(const float* __restrict__ in, const float* __restrict__ wt,
                                   const float* __restrict__ bias, float* __restrict__ out,
                                   int B, int Hin, int Win, int Cin,
                                   int Hout, int Wout, int Cout,
                                   int stride, int dil, int pad, int relu, int addTo) {
    long long idx = (long long)blockIdx.x * blockDim.x + threadIdx.x;
    long long total = (long long)B * Hout * Wout * Cout;
    if (idx >= total) return;
    int co = (int)(idx % Cout);
    long long pix = idx / Cout;
    int ox = (int)(pix % Wout);
    long long t = pix / Wout;
    int oy = (int)(t % Hout);
    int b  = (int)(t / Hout);
    float s = bias[co];
    for (int ky = 0; ky < 3; ++ky) {
        int iy = oy * stride - pad + ky * dil;
        if (iy < 0 || iy >= Hin) continue;
        for (int kx = 0; kx < 3; ++kx) {
            int ix = ox * stride - pad + kx * dil;
            if (ix < 0 || ix >= Win) continue;
            const float* ip = &in[(((size_t)b * Hin + iy) * Win + ix) * Cin];
            const float* wp = &wt[(size_t)(ky * 3 + kx) * Cin * Cout + co];
            for (int ci = 0; ci < Cin; ++ci)
                s += ip[ci] * wp[(size_t)ci * Cout];
        }
    }
    if (relu) s = s > 0.0f ? s : LRELU_NEG * s;
    if (addTo) out[idx] += s; else out[idx] = s;
}

// ---------------------------------------------------------------------------
// 4x4 stride-2 transposed conv (SAME, out = 2*in). Gather form, pad = 1.
// ---------------------------------------------------------------------------
__global__ void pwc_deconv4x4(const float* __restrict__ in, const float* __restrict__ wt,
                              const float* __restrict__ bias, float* __restrict__ out,
                              int B, int Hin, int Win, int Cin, int Cout) {
    int Hout = Hin * 2, Wout = Win * 2;
    long long idx = (long long)blockIdx.x * blockDim.x + threadIdx.x;
    long long total = (long long)B * Hout * Wout * Cout;
    if (idx >= total) return;
    int co = (int)(idx % Cout);
    long long pix = idx / Cout;
    int ox = (int)(pix % Wout);
    long long t = pix / Wout;
    int oy = (int)(t % Hout);
    int b  = (int)(t / Hout);
    float s = bias[co];
    for (int ky = 0; ky < 4; ++ky) {
        int ty = oy + 1 - ky;
        if (ty < 0 || (ty & 1)) continue;
        int iy = ty >> 1;
        if (iy >= Hin) continue;
        for (int kx = 0; kx < 4; ++kx) {
            int tx = ox + 1 - kx;
            if (tx < 0 || (tx & 1)) continue;
            int ix = tx >> 1;
            if (ix >= Win) continue;
            const float* ip = &in[(((size_t)b * Hin + iy) * Win + ix) * Cin];
            const float* wp = &wt[(size_t)(ky * 4 + kx) * Cin * Cout + co];
            for (int ci = 0; ci < Cin; ++ci)
                s += ip[ci] * wp[(size_t)ci * Cout];
        }
    }
    out[idx] = s;
}

// ---------------------------------------------------------------------------
// Bilinear backward warp with flow scaling.
// ---------------------------------------------------------------------------
__global__ void pwc_warp(const float* __restrict__ feat, const float* __restrict__ flow,
                         float scale, float* __restrict__ out,
                         int B, int H, int W, int C) {
    long long idx = (long long)blockIdx.x * blockDim.x + threadIdx.x;
    long long total = (long long)B * H * W * C;
    if (idx >= total) return;
    int c = (int)(idx % C);
    long long pix = idx / C;
    int x = (int)(pix % W);
    long long t = pix / W;
    int y = (int)(t % H);
    int b = (int)(t / H);
    float fx = (float)x + flow[pix * 2 + 0] * scale;
    float fy = (float)y + flow[pix * 2 + 1] * scale;
    float x0f = floorf(fx), y0f = floorf(fy);
    float wx = fx - x0f, wy = fy - y0f;
    int x0 = (int)x0f; x0 = x0 < 0 ? 0 : (x0 > W - 1 ? W - 1 : x0);
    int y0 = (int)y0f; y0 = y0 < 0 ? 0 : (y0 > H - 1 ? H - 1 : y0);
    int x1 = x0 + 1 > W - 1 ? W - 1 : x0 + 1;
    int y1 = y0 + 1 > H - 1 ? H - 1 : y0 + 1;
    size_t base = (size_t)b * H * W;
    float f00 = feat[(base + (size_t)y0 * W + x0) * C + c];
    float f01 = feat[(base + (size_t)y0 * W + x1) * C + c];
    float f10 = feat[(base + (size_t)y1 * W + x0) * C + c];
    float f11 = feat[(base + (size_t)y1 * W + x1) * C + c];
    out[idx] = f00 * (1.0f - wx) * (1.0f - wy) + f01 * wx * (1.0f - wy)
             + f10 * (1.0f - wx) * wy + f11 * wx * wy;
}

// ---------------------------------------------------------------------------
// 81-displacement cost volume (mean over C, leaky relu), writes into a
// channel-interleaved destination of Ctot channels at channel offset Coff.
// ---------------------------------------------------------------------------
__global__ void pwc_cost_volume(const float* __restrict__ c1, const float* __restrict__ c2,
                                float* __restrict__ dst,
                                int B, int H, int W, int C, int Ctot, int Coff) {
    long long idx = (long long)blockIdx.x * blockDim.x + threadIdx.x;
    long long total = (long long)B * H * W * 81;
    if (idx >= total) return;
    int disp = (int)(idx % 81);
    long long pix = idx / 81;
    int x = (int)(pix % W);
    long long t = pix / W;
    int y = (int)(t % H);
    int b = (int)(t / H);
    int dy = disp / 9 - 4;
    int dx = disp % 9 - 4;
    int y2 = y + dy, x2 = x + dx;
    float s = 0.0f;
    if (y2 >= 0 && y2 < H && x2 >= 0 && x2 < W) {
        const float* p1 = &c1[pix * C];
        const float* p2 = &c2[(((size_t)b * H + y2) * W + x2) * C];
        for (int c = 0; c < C; ++c) s += p1[c] * p2[c];
    }
    s /= (float)C;
    s = s > 0.0f ? s : LRELU_NEG * s;
    dst[pix * Ctot + Coff + disp] = s;
}

// ---------------------------------------------------------------------------
// Channel-pack: copy src (Cs channels) into dst (Ct channels) at offset Coff.
// ---------------------------------------------------------------------------
__global__ void pwc_pack(const float* __restrict__ src, int Cs,
                         float* __restrict__ dst, int Ct, int Coff, long long total) {
    long long idx = (long long)blockIdx.x * blockDim.x + threadIdx.x;
    if (idx >= total) return;
    int c = (int)(idx % Cs);
    long long pix = idx / Cs;
    dst[pix * Ct + Coff + c] = src[idx];
}

// ---------------------------------------------------------------------------
// Final bilinear x4 upsample, scaled by 20 (half-pixel centers, edge clamp).
// ---------------------------------------------------------------------------
__global__ void pwc_upsample4x(const float* __restrict__ in, float* __restrict__ out,
                               int B, int Hin, int Win, int C, float scale) {
    int Hout = Hin * 4, Wout = Win * 4;
    long long idx = (long long)blockIdx.x * blockDim.x + threadIdx.x;
    long long total = (long long)B * Hout * Wout * C;
    if (idx >= total) return;
    int c = (int)(idx % C);
    long long pix = idx / C;
    int ox = (int)(pix % Wout);
    long long t = pix / Wout;
    int oy = (int)(t % Hout);
    int b  = (int)(t / Hout);
    float sy = ((float)oy + 0.5f) * 0.25f - 0.5f;
    float sx = ((float)ox + 0.5f) * 0.25f - 0.5f;
    float y0f = floorf(sy), x0f = floorf(sx);
    float wy = sy - y0f, wx = sx - x0f;
    int y0 = (int)y0f; int x0 = (int)x0f;
    int y0c = y0 < 0 ? 0 : (y0 > Hin - 1 ? Hin - 1 : y0);
    int x0c = x0 < 0 ? 0 : (x0 > Win - 1 ? Win - 1 : x0);
    int y1c = y0 + 1 < 0 ? 0 : (y0 + 1 > Hin - 1 ? Hin - 1 : y0 + 1);
    int x1c = x0 + 1 < 0 ? 0 : (x0 + 1 > Win - 1 ? Win - 1 : x0 + 1);
    size_t base = (size_t)b * Hin * Win;
    float f00 = in[(base + (size_t)y0c * Win + x0c) * C + c];
    float f01 = in[(base + (size_t)y0c * Win + x1c) * C + c];
    float f10 = in[(base + (size_t)y1c * Win + x0c) * C + c];
    float f11 = in[(base + (size_t)y1c * Win + x1c) * C + c];
    float v = f00 * (1 - wx) * (1 - wy) + f01 * wx * (1 - wy)
            + f10 * (1 - wx) * wy + f11 * wx * wy;
    out[idx] = v * scale;
}

// ===========================================================================
// Host orchestration
// ===========================================================================
extern "C" void kernel_launch(void* const* d_in, const int* in_sizes, int n_in,
                              void* d_out, int out_size, void* d_ws, size_t ws_size,
                              hipStream_t stream) {
    (void)in_sizes; (void)n_in; (void)out_size; (void)ws_size;

    const float* x1 = (const float*)d_in[0];
    const float* x2 = (const float*)d_in[1];

    // ---- parameter index walk (setup_inputs dict order) ----
    int idx = 2;
    int exta_w[6], exta_b[6], extb_w[6], extb_b[6];
    for (int l = 0; l < 6; ++l) { exta_w[l]=idx++; exta_b[l]=idx++; extb_w[l]=idx++; extb_b[l]=idx++; }
    int est_cw[7][6], est_cb[7][6], est_fw[7], est_fb[7];
    int est_ufw[7], est_ufb[7], est_uffw[7], est_uffb[7];
    const int lvls[5] = {2, 3, 4, 5, 6};
    for (int t = 0; t < 5; ++t) {
        int lvl = lvls[t];
        for (int i = 1; i <= 5; ++i) { est_cw[lvl][i]=idx++; est_cb[lvl][i]=idx++; }
        est_fw[lvl]=idx++; est_fb[lvl]=idx++;
        if (lvl != 2) { est_ufw[lvl]=idx++; est_ufb[lvl]=idx++; est_uffw[lvl]=idx++; est_uffb[lvl]=idx++; }
    }
    int ctxw[7], ctxb[7];
    for (int i = 1; i <= 6; ++i) { ctxw[i]=idx++; ctxb[i]=idx++; }
    int ctxf_w = idx++, ctxf_b = idx++;

    auto P = [&](int i) -> const float* { return (const float*)d_in[i]; };

    // ---- workspace bump allocator ----
    char* wsb = (char*)d_ws;
    size_t wsoff = 0;
    auto allocB = [&](size_t bytes) -> void* {
        void* p = wsb + wsoff;
        wsoff += (bytes + 255) & ~(size_t)255;
        return p;
    };
    auto allocF = [&](size_t n) -> float* { return (float*)allocB(n * 4); };
    auto allocH = [&](size_t n) -> _Float16* { return (_Float16*)allocB(n * 2); };

    // ---- fp16 weight conversion ----
    auto cvt = [&](int pidx, int n) -> const _Float16* {
        _Float16* dst = allocH((size_t)n);
        pwc_f32_to_f16<<<dim3((n + 255) / 256), dim3(256), 0, stream>>>(P(pidx), dst, n);
        return dst;
    };

    // ---- wmma conv launcher ----
    auto convW = [&](const float* in, const _Float16* wt, const float* bias, float* out,
                     int B, int Hin, int Win, int Cin, int Hout, int Wout, int Cout,
                     int strd, int dil, int pad, int relu) {
        dim3 grid((Wout + 31) / 32, (Cout + 31) / 32, B * Hout);
        pwc_conv3x3_wmma<<<grid, dim3(128), 0, stream>>>(
            in, wt, bias, out, B, Hin, Win, Cin, Hout, Wout, Cout, strd, dil, pad, relu);
    };
    auto grid1d = [](long long n) { return dim3((unsigned)((n + 255) / 256)); };

    const int Bn = 4;
    const int EXT_F[6] = {16, 32, 64, 96, 128, 196};
    const int EST_F[5] = {128, 128, 96, 64, 32};
    const int CTXF[6]  = {128, 128, 128, 96, 64, 32};
    const int CTXD[6]  = {1, 2, 4, 8, 16, 1};
    int est_in[7]; est_in[6]=277; est_in[5]=213; est_in[4]=181; est_in[3]=149; est_in[2]=117;

    // ---- convert all WMMA weights ----
    const _Float16 *wExtA[6], *wExtB[6];
    { int cin = 3;
      for (int l = 0; l < 6; ++l) {
          wExtA[l] = cvt(exta_w[l], 9 * cin * EXT_F[l]);
          wExtB[l] = cvt(extb_w[l], 9 * EXT_F[l] * EXT_F[l]);
          cin = EXT_F[l];
      } }
    const _Float16* wEst[7][6];
    for (int t = 0; t < 5; ++t) {
        int lvl = lvls[t], cin = est_in[lvl];
        for (int i = 1; i <= 5; ++i) { wEst[lvl][i] = cvt(est_cw[lvl][i], 9 * cin * EST_F[i-1]); cin = EST_F[i-1]; }
    }
    const _Float16* wCtx[7];
    { int cin = 34;
      for (int i = 1; i <= 6; ++i) { wCtx[i] = cvt(ctxw[i], 9 * cin * CTXF[i-1]); cin = CTXF[i-1]; } }

    // ---- feature pyramid extraction ----
    float* feat[2][6];
    float* tmp = allocF((size_t)Bn * 192 * 384 * 16);
    for (int im = 0; im < 2; ++im) {
        const float* cur = im ? x2 : x1;
        int cin = 3, Hin = 384, Win = 768;
        for (int l = 0; l < 6; ++l) {
            int Hout = Hin / 2, Wout = Win / 2, f = EXT_F[l];
            feat[im][l] = allocF((size_t)Bn * Hout * Wout * f);
            convW(cur, wExtA[l], P(exta_b[l]), tmp, Bn, Hin, Win, cin, Hout, Wout, f, 2, 1, 0, 1);
            convW(tmp, wExtB[l], P(extb_b[l]), feat[im][l], Bn, Hout, Wout, f, Hout, Wout, f, 1, 1, 1, 1);
            cur = feat[im][l]; cin = f; Hin = Hout; Win = Wout;
        }
    }

    // ---- pipeline scratch ----
    const size_t npix2 = (size_t)Bn * 96 * 192;
    float* warped = allocF(npix2 * 32);
    float* concat = allocF(npix2 * 117);
    float* ping   = allocF(npix2 * 128);
    float* pong   = allocF(npix2 * 128);
    float* upflow = allocF(npix2 * 2);
    float* upfeat = allocF(npix2 * 2);

    // ---- output slices ----
    float* out = (float*)d_out;
    float* o_flow = out;                 // 4x384x768x2
    float* o_f2 = out + (size_t)2359296; // 4x 96x192x2
    float* o_f3 = o_f2 + 147456;         // 4x 48x 96x2
    float* o_f4 = o_f3 + 36864;          // 4x 24x 48x2
    float* o_f5 = o_f4 + 9216;           // 4x 12x 24x2
    float* o_f6 = o_f5 + 2304;           // 4x  6x 12x2

    // ---- estimator core (5 wmma convs + direct flow conv) ----
    auto run_est = [&](int lvl, int H, int W, float* cin_buf, float* flow_out) -> float* {
        long long npix = (long long)Bn * H * W;
        float* bufs[2] = {ping, pong};
        float* a = cin_buf;
        int cin = est_in[lvl], pi = 0;
        for (int i = 1; i <= 5; ++i) {
            float* dst = bufs[pi]; pi ^= 1;
            convW(a, wEst[lvl][i], P(est_cb[lvl][i]), dst, Bn, H, W, cin, H, W, EST_F[i-1], 1, 1, 1, 1);
            a = dst; cin = EST_F[i-1];
        }
        pwc_conv3x3_direct<<<grid1d(npix * 2), dim3(256), 0, stream>>>(
            a, P(est_fw[lvl]), P(est_fb[lvl]), flow_out, Bn, H, W, 32, H, W, 2, 1, 1, 1, 0, 0);
        return a;  // 32-ch features
    };

    // ---- level 6 ----
    {
        int H = 6, W = 12, C = 196;
        long long npix = (long long)Bn * H * W;
        pwc_cost_volume<<<grid1d(npix * 81), dim3(256), 0, stream>>>(
            feat[0][5], feat[1][5], concat, Bn, H, W, C, est_in[6], 0);
        pwc_pack<<<grid1d(npix * C), dim3(256), 0, stream>>>(
            feat[0][5], C, concat, est_in[6], 81, npix * C);
        float* xf = run_est(6, H, W, concat, o_f6);
        pwc_deconv4x4<<<grid1d(npix * 4 * 2), dim3(256), 0, stream>>>(
            o_f6, P(est_ufw[6]), P(est_ufb[6]), upflow, Bn, H, W, 2, 2);
        pwc_deconv4x4<<<grid1d(npix * 4 * 2), dim3(256), 0, stream>>>(
            xf, P(est_uffw[6]), P(est_uffb[6]), upfeat, Bn, H, W, 32, 2);
    }

    // ---- levels 5..2 ----
    float scales[7]; scales[5] = 0.625f; scales[4] = 1.25f; scales[3] = 2.5f; scales[2] = 5.0f;
    float* flow_outs[7]; flow_outs[5] = o_f5; flow_outs[4] = o_f4; flow_outs[3] = o_f3; flow_outs[2] = o_f2;
    int CH[7]; CH[2] = 32; CH[3] = 64; CH[4] = 96; CH[5] = 128;
    float* xf2 = nullptr;
    for (int lvl = 5; lvl >= 2; --lvl) {
        int H = 384 >> lvl, W = 768 >> lvl, C = CH[lvl];
        long long npix = (long long)Bn * H * W;
        pwc_warp<<<grid1d(npix * C), dim3(256), 0, stream>>>(
            feat[1][lvl - 1], upflow, scales[lvl], warped, Bn, H, W, C);
        pwc_cost_volume<<<grid1d(npix * 81), dim3(256), 0, stream>>>(
            feat[0][lvl - 1], warped, concat, Bn, H, W, C, est_in[lvl], 0);
        pwc_pack<<<grid1d(npix * C), dim3(256), 0, stream>>>(
            feat[0][lvl - 1], C, concat, est_in[lvl], 81, npix * C);
        pwc_pack<<<grid1d(npix * 2), dim3(256), 0, stream>>>(
            upflow, 2, concat, est_in[lvl], 81 + C, npix * 2);
        pwc_pack<<<grid1d(npix * 2), dim3(256), 0, stream>>>(
            upfeat, 2, concat, est_in[lvl], 81 + C + 2, npix * 2);
        float* xf = run_est(lvl, H, W, concat, flow_outs[lvl]);
        if (lvl != 2) {
            pwc_deconv4x4<<<grid1d(npix * 4 * 2), dim3(256), 0, stream>>>(
                flow_outs[lvl], P(est_ufw[lvl]), P(est_ufb[lvl]), upflow, Bn, H, W, 2, 2);
            pwc_deconv4x4<<<grid1d(npix * 4 * 2), dim3(256), 0, stream>>>(
                xf, P(est_uffw[lvl]), P(est_uffb[lvl]), upfeat, Bn, H, W, 32, 2);
        } else {
            xf2 = xf;
        }
    }

    // ---- context network + residual flow refinement at level 2 ----
    {
        int H = 96, W = 192;
        long long npix = (long long)Bn * H * W;
        pwc_pack<<<grid1d(npix * 2), dim3(256), 0, stream>>>(o_f2, 2, concat, 34, 0, npix * 2);
        pwc_pack<<<grid1d(npix * 32), dim3(256), 0, stream>>>(xf2, 32, concat, 34, 2, npix * 32);
        float* bufs[2] = {pong, ping};   // start with pong: xf2 may alias ping (already packed)
        float* a = concat;
        int cin = 34, pi = 0;
        for (int i = 1; i <= 6; ++i) {
            float* dst = bufs[pi]; pi ^= 1;
            int d = CTXD[i - 1];
            convW(a, wCtx[i], P(ctxb[i]), dst, Bn, H, W, cin, H, W, CTXF[i-1], 1, d, d, 1);
            a = dst; cin = CTXF[i-1];
        }
        pwc_conv3x3_direct<<<grid1d(npix * 2), dim3(256), 0, stream>>>(
            a, P(ctxf_w), P(ctxf_b), o_f2, Bn, H, W, 32, H, W, 2, 1, 1, 1, 0, 1 /*addTo*/);
        pwc_upsample4x<<<grid1d((long long)Bn * 384 * 768 * 2), dim3(256), 0, stream>>>(
            o_f2, o_flow, Bn, 96, 192, 2, 20.0f);
    }
}
